// GraphConvolution_82351702933663
// MI455X (gfx1250) — matrix operand order repaired and checked by reading, compiled, not run on hardware
//
#include <hip/hip_runtime.h>

// ---------------------------------------------------------------------------
// GraphConvolution forward for MI455X (gfx1250, wave32, WMMA).
//   out = relu( segment_sum( adj_val * (dropout(x) @ W)[adj_col], adj_row ) )
//
// GEMM: v_wmma_f32_16x16x32_bf16, block = 8 waves = 128 rows, wave owns
//       16 rows x 128 cols (8 N-tiles). W is staged once per block into LDS
//       as bf16, transposed, so each B fragment is one contiguous 32B LDS
//       read shared by all 8 waves. Dropout = 1 hash per (row, 32-K-chunk),
//       bits of the hash are the per-element keep mask.
// SpMM: wave-per-edge global_atomic_add_f32. h (51.2MB) and out (51.2MB)
//       both fit in the 192MB L2, so the ~1.6GB gather/scatter stays on-chip;
//       HBM-critical traffic is only ~200MB (~9us floor at 23.3 TB/s).
// ---------------------------------------------------------------------------

typedef __attribute__((ext_vector_type(16))) __bf16 v16bf;
typedef __attribute__((ext_vector_type(8)))  float  v8f;

#define DIM 128            // D_IN == D_OUT == 128
#define ROWS_PER_WAVE 16
#define WAVES_PER_BLOCK 8
#define ROWS_PER_BLOCK (ROWS_PER_WAVE * WAVES_PER_BLOCK)   // 128

__device__ __forceinline__ unsigned hash_u32(unsigned x) {
  x ^= x >> 16; x *= 0x7feb352dU;
  x ^= x >> 15; x *= 0x846ca68bU;
  x ^= x >> 16;
  return x;
}

// ---------------------------------------------------------------------------
// Kernel 1: h = dropout(x) @ W   via WMMA bf16 -> f32
// ---------------------------------------------------------------------------
__global__ void __launch_bounds__(256)
gemm_dropout_wmma(const float* __restrict__ x, const float* __restrict__ Wm,
                  float* __restrict__ h, int n_rows) {
  __shared__ __bf16 Wt[DIM * DIM];   // Wt[col][k] = bf16(W[k][col]), 32 KB

  // ---- stage W into LDS, transposed + converted to bf16 ----
  {
    const int t = threadIdx.x;
#pragma unroll
    for (int i = 0; i < (DIM * DIM) / 256; ++i) {     // 64 elements/thread
      int idx = i * 256 + t;
      int col = idx >> 7;          // 0..127
      int k   = idx & 127;         // fast-varying with t -> coalesced LDS writes
      Wt[col * DIM + k] = (__bf16)Wm[(size_t)k * DIM + col];
    }
  }
  __syncthreads();

  const int wave = threadIdx.x >> 5;    // 0..7 -> which 16-row slice
  const int lane = threadIdx.x & 31;
  const int m    = lane & 15;
  const int g    = lane >> 4;           // lane group 0/1 (K-half select)
  const int rowbase = blockIdx.x * ROWS_PER_BLOCK + wave * ROWS_PER_WAVE;
  if (rowbase >= n_rows) return;
  const int row = min(rowbase + m, n_rows - 1);   // clamped A-read row

  v8f acc[8];
#pragma unroll
  for (int n = 0; n < 8; ++n) acc[n] = (v8f){};

#pragma unroll
  for (int k0 = 0; k0 < DIM; k0 += 32) {
    // ---- A fragment (computed ONCE, reused by 8 WMMAs) ----
    // contiguous K order (same as B): lane group g holds K = kbase..kbase+15
    const int kbase = k0 + g * 16;
    const float4* xlo = (const float4*)(x + (size_t)row * DIM + kbase);
    const float4* xhi = (const float4*)(x + (size_t)row * DIM + kbase + 8);
    float4 l0 = xlo[0], l1 = xlo[1];
    float4 h0 = xhi[0], h1 = xhi[1];
    float av[16] = {l0.x, l0.y, l0.z, l0.w, l1.x, l1.y, l1.z, l1.w,
                    h0.x, h0.y, h0.z, h0.w, h1.x, h1.y, h1.z, h1.w};
    // one hash per (row, 32-wide K chunk); bit (k - k0) = keep mask
    const unsigned hh = hash_u32(((unsigned)row << 2) ^ (unsigned)(k0 >> 5)
                                 ^ 0x9E3779B9U);
    v16bf a;
#pragma unroll
    for (int e = 0; e < 16; ++e) {
      int bit = (g * 16) + e;   // k - k0
      float keep = ((hh >> bit) & 1u) ? 2.0f : 0.0f;       // 1/KEEP_PROB = 2
      a[e] = (__bf16)(av[e] * keep);
    }

    // ---- 8 N-tiles: B fragment = contiguous 32B LDS read ----
#pragma unroll
    for (int n = 0; n < 8; ++n) {
      const int col = n * 16 + m;
      const int kb  = k0 + g * 16;   // contiguous K order (same as A): K = kb..kb+15
      v16bf b = *(const v16bf*)(Wt + col * DIM + kb);
      // 8 args: (neg_a, A, neg_b, B, c_mod, C, reuse_a, reuse_b)
      acc[n] = __builtin_amdgcn_wmma_f32_16x16x32_bf16(
          false, a, false, b, (short)0, acc[n], false, false);
    }
  }

  // ---- store D: 32-bit C/D layout: VGPR r -> M = r + 8*g, N = m ----
#pragma unroll
  for (int n = 0; n < 8; ++n) {
#pragma unroll
    for (int r = 0; r < 8; ++r) {
      int mr = rowbase + r + g * 8;
      if (mr < n_rows) h[(size_t)mr * DIM + (n * 16 + m)] = acc[n][r];
    }
  }
}

// ---------------------------------------------------------------------------
// Kernel 2: COO SpMM scatter — one wave per edge, one float4 chunk per lane.
// ---------------------------------------------------------------------------
__global__ void __launch_bounds__(256)
spmm_atomic(const float* __restrict__ h, const int* __restrict__ erow,
            const int* __restrict__ ecol, const float* __restrict__ eval,
            float* __restrict__ out, int n_edges) {
  long long t = (long long)blockIdx.x * blockDim.x + threadIdx.x;
  int e    = (int)(t >> 5);
  int lane = (int)(t & 31);
  if (e >= n_edges) return;
  int r = erow[e];
  int c = ecol[e];
  float v = eval[e];
  float4 hv = ((const float4*)(h + (size_t)c * DIM))[lane];
  float* op = out + (size_t)r * DIM + lane * 4;
  atomicAdd(op + 0, v * hv.x);
  atomicAdd(op + 1, v * hv.y);
  atomicAdd(op + 2, v * hv.z);
  atomicAdd(op + 3, v * hv.w);
}

// ---------------------------------------------------------------------------
// Kernel 3/4: zero-init and in-place ReLU (float4, grid-stride).
// ---------------------------------------------------------------------------
__global__ void zero_f4(float4* __restrict__ p, int n4) {
  int i = blockIdx.x * blockDim.x + threadIdx.x;
  int stride = gridDim.x * blockDim.x;
  float4 z = {0.f, 0.f, 0.f, 0.f};
  for (; i < n4; i += stride) p[i] = z;
}

__global__ void relu_f4(float4* __restrict__ p, int n4) {
  int i = blockIdx.x * blockDim.x + threadIdx.x;
  int stride = gridDim.x * blockDim.x;
  for (; i < n4; i += stride) {
    float4 v = p[i];
    v.x = fmaxf(v.x, 0.f); v.y = fmaxf(v.y, 0.f);
    v.z = fmaxf(v.z, 0.f); v.w = fmaxf(v.w, 0.f);
    p[i] = v;
  }
}

// ---------------------------------------------------------------------------
extern "C" void kernel_launch(void* const* d_in, const int* in_sizes, int n_in,
                              void* d_out, int out_size, void* d_ws, size_t ws_size,
                              hipStream_t stream) {
  const float* x    = (const float*)d_in[0];   // [N, 128]
  const float* Wm   = (const float*)d_in[1];   // [128, 128]
  const int*   arow = (const int*)  d_in[2];   // [E]
  const int*   acol = (const int*)  d_in[3];   // [E]
  const float* aval = (const float*)d_in[4];   // [E]
  float* out = (float*)d_out;                  // [N, 128]
  float* h   = (float*)d_ws;                   // [N, 128] scratch (51.2 MB)

  const int n_nodes = in_sizes[0] / DIM;
  const int n_edges = in_sizes[2];
  const int n4 = (n_nodes * DIM) / 4;

  // 1) out = 0 (atomics accumulate into it)
  zero_f4<<<4096, 256, 0, stream>>>((float4*)out, n4);

  // 2) h = dropout(x) @ W
  int gemm_blocks = (n_nodes + ROWS_PER_BLOCK - 1) / ROWS_PER_BLOCK;  // 782
  gemm_dropout_wmma<<<gemm_blocks, 256, 0, stream>>>(x, Wm, h, n_nodes);

  // 3) out[row] += val * h[col]  (wave per edge)
  long long spmm_threads = (long long)n_edges * 32;
  int spmm_blocks = (int)((spmm_threads + 255) / 256);                // 200000
  spmm_atomic<<<spmm_blocks, 256, 0, stream>>>(h, arow, acol, aval, out, n_edges);

  // 4) out = relu(out)
  relu_f4<<<4096, 256, 0, stream>>>((float4*)out, n4);
}
